// Attention_4801773437196
// MI455X (gfx1250) — matrix operand order repaired and checked by reading, not compiled, not run on hardware
//
#include <hip/hip_runtime.h>
#include <cstdint>

#define S_LEN 4096
#define DM    512
#define NH    8
#define DH    64

typedef unsigned short u16;
typedef unsigned int   u32;
typedef __attribute__((ext_vector_type(16))) __bf16 v16bf;
typedef __attribute__((ext_vector_type(16))) u16    v16u;
typedef __attribute__((ext_vector_type(8)))  u16    v8u;
typedef __attribute__((ext_vector_type(8)))  float  v8f;
typedef __attribute__((ext_vector_type(4)))  float  v4f;
typedef __attribute__((ext_vector_type(2)))  u32    v2u32;

// ---------- helpers ----------

__device__ __forceinline__ u16 f2bf(float f) {
  u32 u = __builtin_bit_cast(u32, f);
  u += 0x7FFFu + ((u >> 16) & 1u);   // round-to-nearest-even
  return (u16)(u >> 16);
}

__device__ __forceinline__ v8f wmma_bf16(v16u a, v16u b, v8f c) {
  // v_wmma_f32_16x16x32_bf16: (neg_a, A, neg_b, B, c_mod, C, reuse_a, reuse_b)
  return __builtin_amdgcn_wmma_f32_16x16x32_bf16(
      false, __builtin_bit_cast(v16bf, a),
      false, __builtin_bit_cast(v16bf, b),
      (short)0, c, false, false);
}

// B-fragment: lane holds N=lane%16; elements e=0..15 are K = 16*hi + e,
// contiguous in memory. All call sites are 32B-aligned -> single v16u load.
__device__ __forceinline__ v16u load16(const u16* __restrict__ p) {
  return *(const v16u*)p;
}

// A-fragment: lane holds row M=lane%16; elements e=0..15 -> K = 16*hi + e.
// Two contiguous 16B loads.
__device__ __forceinline__ v16u afrag(const u16* __restrict__ row, int hi) {
  v8u a = *(const v8u*)(row + 16 * hi);
  v8u b = *(const v8u*)(row + 16 * hi + 8);
  v16u r;
#pragma unroll
  for (int e = 0; e < 8; ++e) { r[e] = a[e]; r[8 + e] = b[e]; }
  return r;
}

// Cross-lane max reduction within each 16-lane half (ds_swizzle xor).
#define SWZ_XOR_F(v, mask)                                              \
  __builtin_bit_cast(float, __builtin_amdgcn_ds_swizzle(                \
      __builtin_bit_cast(int, (v)), (((mask) << 10) | 0x1F)))

__device__ __forceinline__ float rmax16(float v) {
  v = fmaxf(v, SWZ_XOR_F(v, 1));
  v = fmaxf(v, SWZ_XOR_F(v, 2));
  v = fmaxf(v, SWZ_XOR_F(v, 4));
  v = fmaxf(v, SWZ_XOR_F(v, 8));
  return v;
}

// ---------- kernel 0: one-shot f32 -> bf16 conversion of x and all weights ----
__device__ __forceinline__ void cvt4(const float* __restrict__ s,
                                     u16* __restrict__ d) {
  v4f f = *(const v4f*)s;
  u32 lo  = (u32)f2bf(f.x) | ((u32)f2bf(f.y) << 16);
  u32 hi2 = (u32)f2bf(f.z) | ((u32)f2bf(f.w) << 16);
  v2u32 pk = {lo, hi2};
  *(v2u32*)d = pk;
}

__global__ __launch_bounds__(256) void k_cvt(
    const float* __restrict__ x,  const float* __restrict__ Wk,
    const float* __restrict__ Wq, const float* __restrict__ Wv,
    const float* __restrict__ Wo,
    u16* __restrict__ xbf, u16* __restrict__ wkb, u16* __restrict__ wqb,
    u16* __restrict__ wvb, u16* __restrict__ wob)
{
  size_t base = ((size_t)blockIdx.x * 256 + threadIdx.x) * 4;
  const size_t NX = (size_t)2 * S_LEN * DM;   // 4,194,304
  const size_t NW = (size_t)NH * DH * DM;     //   262,144 (also = |W_O|)
  if (base < NX) { cvt4(x  + base, xbf + base); return; }
  base -= NX;
  if (base < NW) { cvt4(Wk + base, wkb + base); return; }
  base -= NW;
  if (base < NW) { cvt4(Wq + base, wqb + base); return; }
  base -= NW;
  if (base < NW) { cvt4(Wv + base, wvb + base); return; }
  base -= NW;
  cvt4(Wo + base, wob + base);
}

// ---------- kernel 1: fused Q/K/V projection (bf16 in -> Q,K,V bf16) ----------
// Block = 128 threads (4 waves), one block per (b, head, 16-row p-tile).
// x tile staged into LDS with CDNA5 async global->LDS DMA (ASYNCcnt).
__global__ __launch_bounds__(128) void k_qkv(
    const u16* __restrict__ Xbf, const u16* __restrict__ Wkb,
    const u16* __restrict__ Wqb, const u16* __restrict__ Wvb,
    u16* __restrict__ Qo, u16* __restrict__ Ko, u16* __restrict__ Vto)
{
  __shared__ __align__(32) u16 xs[16 * DM];  // 16 KB bf16 x tile
  const int bid = blockIdx.x;
  const int pt = bid & 255;
  const int ih = (bid >> 8) & 7;
  const int b  = bid >> 11;
  const int p0 = pt << 4;
  const int t  = threadIdx.x;

  // async copy of the 16x512 bf16 x tile: 1024 chunks of 16B, 8 per thread
  const u16* xsrc = Xbf + ((size_t)b * S_LEN + p0) * DM;
#pragma unroll
  for (int j = 0; j < 8; ++j) {
    const int chunk = t + 128 * j;
    const u32 lds = (u32)(uintptr_t)(&xs[chunk * 8]);
    const u16* g = xsrc + chunk * 8;
    asm volatile("global_load_async_to_lds_b128 %0, %1, off"
                 :: "v"(lds), "v"(g) : "memory");
  }
  asm volatile("s_wait_asynccnt 0x0" ::: "memory");
  __syncthreads();

  const int lane = t & 31;
  const int wave = t >> 5;
  const int hi = lane >> 4;
  const int n  = lane & 15;
  const int h0 = wave << 4;            // this wave's h-tile

  const u16* wqr = Wqb + (size_t)ih * DH * DM + (size_t)(h0 + n) * DM;
  const u16* wkr = Wkb + (size_t)ih * DH * DM + (size_t)(h0 + n) * DM;
  const u16* wvr = Wvb + (size_t)ih * DH * DM + (size_t)(h0 + n) * DM;

  const v8f zero = {0, 0, 0, 0, 0, 0, 0, 0};
  v8f cq = zero, ck = zero, cv = zero;
#pragma unroll 4
  for (int kk = 0; kk < 16; ++kk) {
    const int d0 = kk * 32;
    v16u a = afrag(&xs[n * DM + d0], hi);       // shared A (x tile)
    cq = wmma_bf16(a, load16(wqr + d0 + 16 * hi), cq);
    ck = wmma_bf16(a, load16(wkr + d0 + 16 * hi), ck);
    cv = wmma_bf16(a, load16(wvr + d0 + 16 * hi), cv);
  }

  const size_t head = (size_t)b * NH + ih;
  u16* qdst = Qo + (head * S_LEN + p0) * DH + h0 + n;
  u16* kdst = Ko + (head * S_LEN + p0) * DH + h0 + n;
#pragma unroll
  for (int r = 0; r < 8; ++r) {                 // C: M = r + 8*hi, N = n
    const int M = r + 8 * hi;
    qdst[(size_t)M * DH] = f2bf(cq[r]);
    kdst[(size_t)M * DH] = f2bf(ck[r]);
  }
  v8u vp;                                       // V stored transposed [h][p]:
#pragma unroll                                  // lane's 8 rows are 8 consecutive p
  for (int r = 0; r < 8; ++r) vp[r] = f2bf(cv[r]);
  *(v8u*)(Vto + (head * DH + h0 + n) * S_LEN + p0 + 8 * hi) = vp;
}

// ---------- kernel 2: causal flash attention (bf16 in, bf16 z out) ----------
// Block = 128 threads (4 waves); each wave owns a 32-row q-tile (two 16-row
// subtiles A/B) so every K/V fragment load feeds 2x the WMMAs. Row-sums (l)
// are accumulated on the matrix pipe via a constant ones B-fragment.
__device__ __forceinline__ void softmax_update(
    v8f s0, v8f s1, int qbase, int p0, int hi, int n,
    float* __restrict__ m, float* __restrict__ alpha,
    u16* __restrict__ prow)
{
#pragma unroll
  for (int r = 0; r < 8; ++r) {
    const int qg = qbase + r + 8 * hi;          // global query row of this element
    const bool k0 = (p0 + n)      <= qg;        // causal mask
    const bool k1 = (p0 + 16 + n) <= qg;
    const float t0 = k0 ? s0[r] * 0.125f : -3.0e38f;   // scale = 1/sqrt(64)
    const float t1 = k1 ? s1[r] * 0.125f : -3.0e38f;
    const float mr = rmax16(fmaxf(t0, t1));
    const float mn = fmaxf(m[r], mr);
    alpha[r] = __expf(m[r] - mn);
    m[r] = mn;
    const float e0 = k0 ? __expf(t0 - mn) : 0.0f;
    const float e1 = k1 ? __expf(t1 - mn) : 0.0f;
    const int M = r + 8 * hi;                   // C-layout row -> LDS [M][col]
    prow[M * 32 + n]      = f2bf(e0);
    prow[M * 32 + 16 + n] = f2bf(e1);
  }
}

__global__ __launch_bounds__(128) void k_attn(
    const u16* __restrict__ Q, const u16* __restrict__ K,
    const u16* __restrict__ Vt, u16* __restrict__ Z)
{
  __shared__ __align__(32) u16 ps[4][32 * 32];  // per-wave 32x32 P staging (8 KB)
  const int bid = blockIdx.x;
  const int qt = bid & 31;
  const int ih = (bid >> 5) & 7;
  const int b  = bid >> 8;
  const int lane = threadIdx.x & 31;
  const int wave = threadIdx.x >> 5;
  const int q0 = qt * 128 + wave * 32;          // 32 query rows per wave
  const int hi = lane >> 4;
  const int n  = lane & 15;

  const size_t head = (size_t)b * NH + ih;
  const u16* Qh = Q  + head * S_LEN * DH;
  const u16* Kh = K  + head * S_LEN * DH;
  const u16* Vh = Vt + head * DH * (size_t)S_LEN;

  // Q A-fragments: subtile A = rows q0..q0+15, subtile B = rows q0+16..q0+31
  const u16* qrowA = Qh + (size_t)(q0 + n) * DH;
  const u16* qrowB = qrowA + 16 * DH;
  const v16u qa0 = afrag(qrowA, hi);
  const v16u qa1 = afrag(qrowA + 32, hi);
  const v16u qb0 = afrag(qrowB, hi);
  const v16u qb1 = afrag(qrowB + 32, hi);

  v16u ones;                                    // bf16 1.0 B-fragment (row sums)
#pragma unroll
  for (int e = 0; e < 16; ++e) ones[e] = 0x3F80;

  const v8f zero = {0, 0, 0, 0, 0, 0, 0, 0};
  v8f aA0 = zero, aA1 = zero, aA2 = zero, aA3 = zero, alA = zero;
  v8f aB0 = zero, aB1 = zero, aB2 = zero, aB3 = zero, alB = zero;
  float mA[8], mB[8], alphaA[8], alphaB[8];
#pragma unroll
  for (int r = 0; r < 8; ++r) { mA[r] = -3.0e38f; mB[r] = -3.0e38f; }

  u16* myp = &ps[wave][0];
  const int nsteps = (q0 >> 5) + 1;             // keys <= q0+31
  for (int s = 0; s < nsteps; ++s) {
    const int p0 = s * 32;
    const u16* krow0 = Kh + (size_t)(p0 + n) * DH;
    const u16* krow1 = krow0 + 16 * DH;
    __builtin_prefetch(krow0 + 32 * DH, 0, 1);  // global_prefetch_b8: next K tile
    __builtin_prefetch(krow1 + 32 * DH, 0, 1);

    // 4 K B-fragments, each reused by both q-subtiles
    const v16u kb0lo = load16(krow0 + 16 * hi);
    const v16u kb0hi = load16(krow0 + 32 + 16 * hi);
    const v16u kb1lo = load16(krow1 + 16 * hi);
    const v16u kb1hi = load16(krow1 + 32 + 16 * hi);

    // S = Q * K^T : 4 score tiles, K-dim = 64 -> 8 WMMAs
    v8f sA0 = wmma_bf16(qa1, kb0hi, wmma_bf16(qa0, kb0lo, zero));
    v8f sA1 = wmma_bf16(qa1, kb1hi, wmma_bf16(qa0, kb1lo, zero));
    v8f sB0 = wmma_bf16(qb1, kb0hi, wmma_bf16(qb0, kb0lo, zero));
    v8f sB1 = wmma_bf16(qb1, kb1hi, wmma_bf16(qb0, kb1lo, zero));

    // online softmax per subtile; P tiles written to LDS in C-layout order
    softmax_update(sA0, sA1, q0,      p0, hi, n, mA, alphaA, myp);
    softmax_update(sB0, sB1, q0 + 16, p0, hi, n, mB, alphaB, myp + 16 * 32);

#pragma unroll
    for (int r = 0; r < 8; ++r) {               // rescale running accumulators
      aA0[r] *= alphaA[r]; aA1[r] *= alphaA[r]; aA2[r] *= alphaA[r];
      aA3[r] *= alphaA[r]; alA[r] *= alphaA[r];
      aB0[r] *= alphaB[r]; aB1[r] *= alphaB[r]; aB2[r] *= alphaB[r];
      aB3[r] *= alphaB[r]; alB[r] *= alphaB[r];
    }
    asm volatile("s_wait_dscnt 0x0" ::: "memory");  // P stores -> A-frag reads
    const v16u pa = afrag(myp + n * 32, hi);        // P subtile A as A-fragment
    const v16u pb = afrag(myp + (16 + n) * 32, hi); // P subtile B as A-fragment

    // 4 V B-fragments (transposed V -> contiguous), reused by both subtiles
    const u16* vb = Vh + (size_t)n * S_LEN + p0 + 16 * hi;
    const v16u vb0 = load16(vb);
    const v16u vb1 = load16(vb + 16 * S_LEN);
    const v16u vb2 = load16(vb + 32 * S_LEN);
    const v16u vb3 = load16(vb + 48 * S_LEN);

    // Z += P*V and l += P*1 : 10 WMMAs
    aA0 = wmma_bf16(pa, vb0, aA0);
    aA1 = wmma_bf16(pa, vb1, aA1);
    aA2 = wmma_bf16(pa, vb2, aA2);
    aA3 = wmma_bf16(pa, vb3, aA3);
    alA = wmma_bf16(pa, ones, alA);
    aB0 = wmma_bf16(pb, vb0, aB0);
    aB1 = wmma_bf16(pb, vb1, aB1);
    aB2 = wmma_bf16(pb, vb2, aB2);
    aB3 = wmma_bf16(pb, vb3, aB3);
    alB = wmma_bf16(pb, ones, alB);
  }

  // epilogue: Z /= l, store bf16 into z_flat [b][q][i*64+h]
  u16* zbA = Z + ((size_t)b * S_LEN + q0) * DM + ih * DH + n;
  u16* zbB = zbA + (size_t)16 * DM;
#pragma unroll
  for (int r = 0; r < 8; ++r) {
    const size_t roff = (size_t)(r + 8 * hi) * DM;
    const float iA = 1.0f / alA[r];             // l broadcast across columns
    zbA[roff]      = f2bf(aA0[r] * iA);
    zbA[roff + 16] = f2bf(aA1[r] * iA);
    zbA[roff + 32] = f2bf(aA2[r] * iA);
    zbA[roff + 48] = f2bf(aA3[r] * iA);
    const float iB = 1.0f / alB[r];
    zbB[roff]      = f2bf(aB0[r] * iB);
    zbB[roff + 16] = f2bf(aB1[r] * iB);
    zbB[roff + 32] = f2bf(aB2[r] * iB);
    zbB[roff + 48] = f2bf(aB3[r] * iB);
  }
}

// ---------- kernel 3: output projection out = z * W_O^T (f32 out) ----------
__global__ __launch_bounds__(128) void k_oproj(
    const u16* __restrict__ Z, const u16* __restrict__ Wob,
    float* __restrict__ out)
{
  const int id   = blockIdx.x * 4 + (threadIdx.x >> 5);  // one 16x16 tile per wave
  const int lane = threadIdx.x & 31;
  const int dt = id & 31;
  const int qt = (id >> 5) & 255;
  const int b  = id >> 13;
  const int q0 = qt << 4;
  const int d0 = dt << 4;
  const int hi = lane >> 4;
  const int n  = lane & 15;

  const u16* zrow = Z   + ((size_t)b * S_LEN + q0 + n) * DM;
  const u16* wrow = Wob + (size_t)(d0 + n) * DM;
  const v8f zero = {0, 0, 0, 0, 0, 0, 0, 0};
  v8f c = zero;
#pragma unroll 4
  for (int kk = 0; kk < 16; ++kk) {
    const int f0 = kk * 32;
    v16u a   = afrag(zrow + f0, hi);
    v16u bfr = load16(wrow + f0 + 16 * hi);
    c = wmma_bf16(a, bfr, c);
  }
  float* ob = out + ((size_t)b * S_LEN + q0) * DM + d0 + n;
#pragma unroll
  for (int r = 0; r < 8; ++r)
    ob[(size_t)(r + 8 * hi) * DM] = c[r];
}

// ---------- launch ----------
extern "C" void kernel_launch(void* const* d_in, const int* in_sizes, int n_in,
                              void* d_out, int out_size, void* d_ws, size_t ws_size,
                              hipStream_t stream) {
  (void)in_sizes; (void)n_in; (void)out_size; (void)ws_size;
  const float* x  = (const float*)d_in[0];
  const float* Wk = (const float*)d_in[1];
  const float* Wq = (const float*)d_in[2];
  const float* Wv = (const float*)d_in[3];
  const float* Wo = (const float*)d_in[4];
  float* out = (float*)d_out;

  // workspace layout (all u16/bf16):
  //  xbf | Wk | Wq | Wv | Wo | Q | K | V^T | Z   (42 MiB total)
  u16* ws = (u16*)d_ws;
  const size_t NX  = (size_t)2 * S_LEN * DM;       // 4,194,304
  const size_t NW  = (size_t)NH * DH * DM;         //   262,144
  const size_t BUF = (size_t)2 * NH * S_LEN * DH;  // 4,194,304
  u16* Xbf = ws;
  u16* Wkb = Xbf + NX;
  u16* Wqb = Wkb + NW;
  u16* Wvb = Wqb + NW;
  u16* Wob = Wvb + NW;
  u16* Q   = Wob + NW;
  u16* K   = Q   + BUF;
  u16* Vt  = K   + BUF;
  u16* Z   = Vt  + BUF;

  // (NX + 4*NW) / 4 floats per thread / 256 threads = 5120 blocks exactly
  k_cvt  <<<dim3(5120), dim3(256), 0, stream>>>(x, Wk, Wq, Wv, Wo,
                                                Xbf, Wkb, Wqb, Wvb, Wob);
  k_qkv  <<<dim3(2 * NH * (S_LEN / 16)), dim3(128), 0, stream>>>(
      Xbf, Wkb, Wqb, Wvb, Q, K, Vt);
  k_attn <<<dim3(2 * NH * (S_LEN / 128)), dim3(128), 0, stream>>>(Q, K, Vt, Z);
  k_oproj<<<dim3(2 * (S_LEN / 16) * (DM / 16) / 4), dim3(128), 0, stream>>>(
      Z, Wob, out);
}
